// ComplexDotProductAttention_38792144617705
// MI455X (gfx1250) — matrix operand order repaired and checked by reading, not compiled, not run on hardware
//
#include <hip/hip_runtime.h>

typedef __attribute__((ext_vector_type(16))) _Float16 v16h;
typedef __attribute__((ext_vector_type(8)))  float    v8f;
typedef __attribute__((ext_vector_type(4)))  float    v4f;
typedef __attribute__((ext_vector_type(2)))  float    v2f;

#define WMMA(a, b, c) \
    __builtin_amdgcn_wmma_f32_16x16x32_f16(false, (a), false, (b), (short)0, (c), false, false)

constexpr int BATCH = 4;
constexpr int SEQ   = 4096;   // Q == K length
constexpr int DD    = 64;     // head dim (contraction for QK^T)
constexpr int VD    = 64;     // value dim
constexpr int MT    = 64;     // q rows per block
constexpr int NT    = 32;     // keys per iteration
constexpr int NWAVE = 4;
constexpr int PSTR  = 48;     // halves per P-scratch row: 96B, 32B-aligned rows

// ---------------------------------------------------------------------------
// One-time preconversion: deinterleave K into f16 planes [B][K][D] and
// transpose V into f16 planes [B][V][K].  O(K*D) work, amortized over the
// O(Q*K) main loop; keeps all per-tile fragment loads conversion-free.
// ---------------------------------------------------------------------------
__global__ __launch_bounds__(256)
void preconvert(const float* __restrict__ gk, const float* __restrict__ gv,
                _Float16* __restrict__ krh, _Float16* __restrict__ kih,
                _Float16* __restrict__ vtr, _Float16* __restrict__ vti)
{
    const int idx = blockIdx.x * 256 + threadIdx.x;     // 0 .. B*SEQ*DD-1
    // K planes: idx = (b*SEQ + k)*DD + d   (reads coalesced, writes coalesced)
    {
        v2f f = *(const v2f*)(gk + (size_t)idx * 2);
        krh[idx] = (_Float16)f.x;
        kih[idx] = (_Float16)f.y;
    }
    // V^T planes: idx = (b*VD + v)*SEQ + k  (writes coalesced along k)
    {
        int k  = idx & (SEQ - 1);
        int t  = idx >> 12;          // b*VD + v
        int vv = t & (VD - 1);
        int bb = t >> 6;
        v2f f = *(const v2f*)(gv + ((size_t)((bb * SEQ + k) * VD + vv)) * 2);
        vtr[idx] = (_Float16)f.x;
        vti[idx] = (_Float16)f.y;
    }
}

// ---------------------------------------------------------------------------
// Fused complex flash-attention main kernel
// ---------------------------------------------------------------------------
__global__ __launch_bounds__(NWAVE * 32)
void cattn_fused(const float* __restrict__ gq,
                 const _Float16* __restrict__ krh, const _Float16* __restrict__ kih,
                 const _Float16* __restrict__ vtr, const _Float16* __restrict__ vti,
                 float* __restrict__ gout)
{
    // double-buffered wave-private P scratch (Ar plane + Ai plane per wave)
    __shared__ __attribute__((aligned(32))) _Float16 Psc[2 * NWAVE * 2 * 16 * PSTR];

    const int tid  = threadIdx.x;
    const int wave = tid >> 5;
    const int lane = tid & 31;
    const int r    = lane & 15;   // row/col within 16-lane half
    const int h    = lane >> 4;   // which half of the wave

    const int b     = blockIdx.x >> 6;            // 64 q-tiles per batch
    const int mbase = (blockIdx.x & 63) * MT + wave * 16;

    const float scale = 0.125f;   // 1/sqrt(64)

    const float*    qb  = gq  + (size_t)b * SEQ * DD * 2;
    const _Float16* krb = krh + (size_t)b * SEQ * DD;
    const _Float16* kib = kih + (size_t)b * SEQ * DD;
    const _Float16* vrb = vtr + (size_t)b * VD * SEQ;
    const _Float16* vib = vti + (size_t)b * VD * SEQ;
    float*          ob  = gout + (size_t)b * SEQ * VD * 2;

    // ---- Q fragments in A-matrix layout: 2 chunks of 16x32 (D=64), scale folded ----
    v16h qrF[2], qiF[2], qinF[2];
    {
        const float* qrow = qb + (size_t)(mbase + r) * DD * 2;   // interleaved (re,im)
        #pragma unroll
        for (int c = 0; c < 2; ++c) {
            _Float16 tr[16], ti[16];
            #pragma unroll
            for (int run = 0; run < 2; ++run) {
                const int d0 = c * 32 + h * 16 + run * 8;        // contiguous k order: K=0..15 / 16..31 per half
                const v4f* p = (const v4f*)(qrow + d0 * 2);
                #pragma unroll
                for (int j = 0; j < 4; ++j) {
                    v4f f = p[j];
                    tr[run*8 + j*2 + 0] = (_Float16)(f.x * scale);
                    ti[run*8 + j*2 + 0] = (_Float16)(f.y * scale);
                    tr[run*8 + j*2 + 1] = (_Float16)(f.z * scale);
                    ti[run*8 + j*2 + 1] = (_Float16)(f.w * scale);
                }
            }
            v16h vr, vi;
            #pragma unroll
            for (int j = 0; j < 16; ++j) { vr[j] = tr[j]; vi[j] = ti[j]; }
            qrF[c] = vr; qiF[c] = vi; qinF[c] = -vi;
        }
    }

    // ---- accumulators: Y (16 x 64, complex, f32) + online softmax state ----
    v8f yr[4], yi[4];
    const v8f vz = {};
    #pragma unroll
    for (int t = 0; t < 4; ++t) { yr[t] = vz; yi[t] = vz; }
    float mr[8], lr[8];
    #pragma unroll
    for (int i = 0; i < 8; ++i) { mr[i] = -3.0e38f; lr[i] = 0.0f; }

    // P-scratch write column: identity key order (subtile 1 at +16) so the
    // read-side A-fragment (16 halves per lane) is one contiguous 32B LDS load.
    const int q0 = r;

    for (int kt = 0; kt < SEQ / NT; ++kt) {
        // ---- S = Q K^T for two 16-key subtiles (8 WMMAs each); fragments are
        //      single aligned 32B loads from the f16 planes ----
        v8f sr[2], si[2];
        #pragma unroll
        for (int s = 0; s < 2; ++s) {
            const size_t krow = (size_t)(kt * NT + s * 16 + r) * DD;
            const int d0 = h * 16;            // contiguous k order, as Q: K=0..15 / 16..31 per half
            v16h kr0 = *(const v16h*)(krb + krow + d0);
            v16h kr1 = *(const v16h*)(krb + krow + 32 + d0);
            v16h ki0 = *(const v16h*)(kib + krow + d0);
            v16h ki1 = *(const v16h*)(kib + krow + 32 + d0);
            __builtin_prefetch(krb + krow + NT * DD, 0, 3);   // next tile (speculative)
            __builtin_prefetch(kib + krow + NT * DD, 0, 3);
            // Sr = Qr Kr^T + Qi Ki^T   (scale already folded into Q)
            sr[s] = WMMA(qiF[1], ki1,
                    WMMA(qiF[0], ki0,
                    WMMA(qrF[1], kr1,
                    WMMA(qrF[0], kr0, vz))));
            // Si = Qr Ki^T - Qi Kr^T
            si[s] = WMMA(qinF[1], kr1,
                    WMMA(qinF[0], kr0,
                    WMMA(qrF[1], ki1,
                    WMMA(qrF[0], ki0, vz))));
        }

        // ---- online norm-softmax: change = exp(norm - m) / (norm + eps)
        //      norm = ss*rsqrt(ss); 1/(norm+eps) ~= rsqrt(ss)  (guarded) ----
        float c0[8], c1[8];
        v8f corrv;
        #pragma unroll
        for (int i = 0; i < 8; ++i) {
            float ss0 = sr[0][i]*sr[0][i] + si[0][i]*si[0][i];
            float ss1 = sr[1][i]*sr[1][i] + si[1][i]*si[1][i];
            float rq0 = __builtin_amdgcn_rsqf(ss0 + 1e-18f);
            float rq1 = __builtin_amdgcn_rsqf(ss1 + 1e-18f);
            float n0  = ss0 * rq0;
            float n1  = ss1 * rq1;
            float tmax = fmaxf(n0, n1);
            #pragma unroll
            for (int mk = 1; mk <= 8; mk <<= 1)
                tmax = fmaxf(tmax, __shfl_xor(tmax, mk, 32));
            float nm = fmaxf(mr[i], tmax);
            float cr = __expf(mr[i] - nm);
            mr[i] = nm;
            float e0 = __expf(n0 - nm), e1 = __expf(n1 - nm);
            float rs = e0 + e1;
            #pragma unroll
            for (int mk = 1; mk <= 8; mk <<= 1)
                rs += __shfl_xor(rs, mk, 32);
            lr[i] = lr[i] * cr + rs;
            c0[i] = e0 * rq0;
            c1[i] = e1 * rq1;
            corrv[i] = cr;
        }
        #pragma unroll
        for (int t = 0; t < 4; ++t) { yr[t] *= corrv; yi[t] *= corrv; }

        // ---- spill Ar/Ai (C layout -> row-major f16, wave-private) ----
        _Float16* Pw = Psc + (((kt & 1) * NWAVE + wave) * 2 * 16) * PSTR;
        #pragma unroll
        for (int i = 0; i < 8; ++i) {
            int row = h * 8 + i;              // C layout: half selects rows 0-7 / 8-15
            Pw[row * PSTR + q0]     = (_Float16)(sr[0][i] * c0[i]);
            Pw[row * PSTR + q0 + 16] = (_Float16)(sr[1][i] * c1[i]);
            Pw[16*PSTR + row * PSTR + q0]     = (_Float16)(si[0][i] * c0[i]);
            Pw[16*PSTR + row * PSTR + q0 + 16] = (_Float16)(si[1][i] * c1[i]);
        }
        __syncthreads();                      // cross-lane visibility of P scratch

        // ---- reload P as A-matrix fragments: one 32B LDS load each ----
        v16h arf  = *(const v16h*)&Pw[r * PSTR + h * 16];
        v16h aif  = *(const v16h*)&Pw[16*PSTR + r * PSTR + h * 16];
        v16h ainf = -aif;

        // ---- Y += A . V  (complex): V^T-plane frags are single 32B loads ----
        #pragma unroll
        for (int t = 0; t < 4; ++t) {
            const _Float16* vcr = vrb + (size_t)(t * 16 + r) * SEQ + kt * NT + h * 16;
            const _Float16* vci = vib + (size_t)(t * 16 + r) * SEQ + kt * NT + h * 16;
            v16h vrf = *(const v16h*)vcr;
            v16h vif = *(const v16h*)vci;
            __builtin_prefetch(vcr + NT, 0, 3);
            __builtin_prefetch(vci + NT, 0, 3);
            yr[t] = WMMA(arf, vrf, WMMA(ainf, vif, yr[t]));  // Ar*Vr - Ai*Vi
            yi[t] = WMMA(arf, vif, WMMA(aif,  vrf, yi[t]));  // Ar*Vi + Ai*Vr
        }
        // no trailing barrier: P scratch is double-buffered on kt&1
    }

    // ---- epilogue: divide by l, write interleaved (re, im) ----
    float invl[8];
    #pragma unroll
    for (int i = 0; i < 8; ++i) invl[i] = __builtin_amdgcn_rcpf(lr[i]);
    #pragma unroll
    for (int t = 0; t < 4; ++t) {
        #pragma unroll
        for (int i = 0; i < 8; ++i) {
            int row = h * 8 + i;
            int col = t * 16 + r;
            v2f o;
            o.x = yr[t][i] * invl[i];
            o.y = yi[t][i] * invl[i];
            *(v2f*)(ob + ((size_t)(mbase + row) * VD + col) * 2) = o;
        }
    }
}

extern "C" void kernel_launch(void* const* d_in, const int* in_sizes, int n_in,
                              void* d_out, int out_size, void* d_ws, size_t ws_size,
                              hipStream_t stream) {
    (void)in_sizes; (void)n_in; (void)out_size; (void)ws_size;  // needs 8 MB of ws
    const float* q = (const float*)d_in[0];
    const float* k = (const float*)d_in[1];
    const float* v = (const float*)d_in[2];
    float* o = (float*)d_out;

    const size_t plane = (size_t)BATCH * SEQ * DD;   // 1 Mi halves per plane
    _Float16* krh = (_Float16*)d_ws;
    _Float16* kih = krh + plane;
    _Float16* vtr = kih + plane;
    _Float16* vti = vtr + plane;

    preconvert<<<dim3((unsigned)(plane / 256)), dim3(256), 0, stream>>>(k, v, krh, kih, vtr, vti);

    dim3 grid(BATCH * (SEQ / MT));       // 256 blocks
    dim3 block(NWAVE * 32);              // 4 waves
    cattn_fused<<<grid, block, 0, stream>>>(q, krh, kih, vtr, vti, o);
}
